// MultiHeadAttention_18957985645158
// MI455X (gfx1250) — matrix operand order repaired and checked by reading, not compiled, not run on hardware
//
#include <hip/hip_runtime.h>
#include <hip/hip_bf16.h>
#include <math.h>
#include <stdint.h>

typedef __attribute__((ext_vector_type(16))) _Float16 v16h;
typedef __attribute__((ext_vector_type(8)))  _Float16 v8h;
typedef __attribute__((ext_vector_type(4)))  _Float16 v4h;
typedef __attribute__((ext_vector_type(8)))  float    v8f;
typedef __attribute__((ext_vector_type(4)))  float    v4f;

#define D_MODEL 1024
#define NHEAD   16
#define HD      64
#define SEQ     2048
#define BATCH   4
#define NQKV    (3 * D_MODEL)       // 3072
#define MTOT    (BATCH * SEQ)       // 8192

// ---------------------------------------------------------------------------
// CDNA5 async copy helpers (ASYNCcnt path): global -> LDS, 16B per lane.
// LDS byte offset = low 32 bits of the generic shared pointer (LDS aperture).
// ---------------------------------------------------------------------------
__device__ __forceinline__ void async_g2l_b128(const _Float16* g, const _Float16* l) {
  unsigned lds = (unsigned)(uintptr_t)l;
  asm volatile("global_load_async_to_lds_b128 %0, %1, off"
               :: "v"(lds), "v"(g) : "memory");
}
__device__ __forceinline__ void wait_async0() {
  asm volatile("s_wait_asynccnt 0x0" ::: "memory");
}

// ---------------------------------------------------------------------------
// WMMA fragment helpers (wave32; A/B k order is this code's contiguous one, C/D per ISA)
// A (16x32 f16): lane m=lane&15, h=lane>>4 holds K = {16h .. 16h+15}
// B (32x16 f16): lane n=lane&15, h=lane>>4 holds K = {16h .. 16h+15} of column n
// C/D (16x16 f32): vgpr j holds row (j + 8*(lane>>4)), col = lane&15
// ---------------------------------------------------------------------------
__device__ __forceinline__ v16h load_a_frag(const _Float16* __restrict__ base,
                                            int ld, int k0, int lane) {
  const int m = lane & 15, hf = lane >> 4;
  const _Float16* p = base + (size_t)m * ld + k0 + 16 * hf;
  v8h lo = *(const v8h*)(p);
  v8h hi = *(const v8h*)(p + 8);
  v16h r;
#pragma unroll
  for (int i = 0; i < 8; ++i) { r[i] = lo[i]; r[i + 8] = hi[i]; }
  return r;
}

// B fragment from an LDS panel: panel row = column n of the output tile,
// `ldp` halves per row (padded). Needs (ldp*2) % 32 == 0 for b128 alignment.
__device__ __forceinline__ v16h lds_b_frag32(const _Float16* base, int t,
                                             int ldp, int lane) {
  const int n = lane & 15, hf = lane >> 4;
  return *(const v16h*)(base + (t * 16 + n) * ldp + 16 * hf);
}
// Same but via two 16B reads (for 16B-aligned pads like 72/40 halves).
__device__ __forceinline__ v16h lds_b_frag16(const _Float16* base, int t,
                                             int ldp, int lane) {
  const int n = lane & 15, hf = lane >> 4;
  const _Float16* p = base + (t * 16 + n) * ldp + 16 * hf;
  v8h lo = *(const v8h*)p;
  v8h hi = *(const v8h*)(p + 8);
  v16h r;
#pragma unroll
  for (int i = 0; i < 8; ++i) { r[i] = lo[i]; r[i + 8] = hi[i]; }
  return r;
}

__device__ __forceinline__ v8f wmma_f16(v16h a, v16h b, v8f c) {
  return __builtin_amdgcn_wmma_f32_16x16x32_f16(false, a, false, b, (short)0, c,
                                                false, false);
}

// ---------------------------------------------------------------------------
// Conversion kernels
// ---------------------------------------------------------------------------
__global__ void __launch_bounds__(256)
k_cvt_f32_f16(const float* __restrict__ src, _Float16* __restrict__ dst, int n) {
  int i = (blockIdx.x * blockDim.x + threadIdx.x) * 4;
  if (i + 3 < n) {
    v4f v = *(const v4f*)(src + i);
    v4h o;
#pragma unroll
    for (int k = 0; k < 4; ++k) o[k] = (_Float16)v[k];
    *(v4h*)(dst + i) = o;
  }
}

// W: [K][N] row-major f32  ->  Wt: [N][K] row-major f16
__global__ void __launch_bounds__(256)
k_cvt_transpose(const float* __restrict__ W, _Float16* __restrict__ Wt,
                int K, int N) {
  int idx = blockIdx.x * blockDim.x + threadIdx.x;
  if (idx < K * N) {
    int nn = idx / K, kk = idx - nn * K;
    Wt[idx] = (_Float16)W[(size_t)kk * N + nn];
  }
}

// ---------------------------------------------------------------------------
// GEMM1: qkv = x @ W_qkv + b_qkv. Block = 128 rows x 64 cols (8 waves x 16
// rows) sharing one async-staged, double-buffered weight panel in LDS.
// Scatter into Q [B,H,S,64], K [B,H,S,64], V transposed [B,H,64,S], all f16.
// ---------------------------------------------------------------------------
#define BP_LD 48   // padded halves per weight-panel row (96B: 32B-aligned)

__global__ void __launch_bounds__(256)
k_gemm_qkv(const _Float16* __restrict__ xh, const _Float16* __restrict__ WqkvT,
           const float* __restrict__ bqkv,
           _Float16* __restrict__ Qh, _Float16* __restrict__ Kh,
           _Float16* __restrict__ Vt) {
  __shared__ _Float16 Bsm[2][64 * BP_LD];
  const int tid = threadIdx.x;
  const int lane = tid & 31, wslot = tid >> 5;
  const int NT = NQKV / 64;                  // 48 col tiles
  const int bm = blockIdx.x / NT;
  const int bn = blockIdx.x - bm * NT;
  const int row0 = bm * 128 + wslot * 16;
  const int col0 = bn * 64;

  const _Float16* gB = WqkvT + (size_t)col0 * D_MODEL;
  const int sr = tid >> 2, sc = (tid & 3) * 8;     // 64 rows x 4 chunks of 16B

  v8f acc[4] = {};
  const _Float16* Arow = xh + (size_t)row0 * D_MODEL;

  async_g2l_b128(gB + (size_t)sr * D_MODEL + sc, &Bsm[0][sr * BP_LD + sc]);
  wait_async0();
  __syncthreads();

  for (int k0 = 0; k0 < D_MODEL; k0 += 32) {
    const int cur = (k0 >> 5) & 1;
    if (k0 + 32 < D_MODEL)                         // prefetch next panel
      async_g2l_b128(gB + (size_t)sr * D_MODEL + (k0 + 32) + sc,
                     &Bsm[cur ^ 1][sr * BP_LD + sc]);
    v16h a = load_a_frag(Arow, D_MODEL, k0, lane);
    const _Float16* Bb = &Bsm[cur][0];
#pragma unroll
    for (int t = 0; t < 4; ++t)
      acc[t] = wmma_f16(a, lds_b_frag32(Bb, t, BP_LD, lane), acc[t]);
    wait_async0();
    __syncthreads();
  }

  const int hf = lane >> 4, n = lane & 15;
#pragma unroll
  for (int t = 0; t < 4; ++t) {
#pragma unroll
    for (int j = 0; j < 8; ++j) {
      int row = row0 + j + 8 * hf;
      int col = col0 + t * 16 + n;
      float v = acc[t][j] + bqkv[col];
      int bi = row / SEQ, s = row - bi * SEQ;
      _Float16 hv = (_Float16)v;
      if (col < D_MODEL) {                       // Q
        int hh = col >> 6, d = col & 63;
        Qh[(((size_t)bi * NHEAD + hh) * SEQ + s) * HD + d] = hv;
      } else if (col < 2 * D_MODEL) {            // K
        int c = col - D_MODEL, hh = c >> 6, d = c & 63;
        Kh[(((size_t)bi * NHEAD + hh) * SEQ + s) * HD + d] = hv;
      } else {                                   // V (transposed)
        int c = col - 2 * D_MODEL, hh = c >> 6, d = c & 63;
        Vt[(((size_t)bi * NHEAD + hh) * HD + d) * SEQ + s] = hv;
      }
    }
  }
}

// ---------------------------------------------------------------------------
// Flash attention. All 8 waves of a block share (b,h); K/V panels for each
// 32-key step are async-staged into double-buffered LDS and shared.
// One wave per 16-query tile, online softmax in f32.
// ---------------------------------------------------------------------------
#define KP_LD 72   // padded halves per K-panel row (144B, 16B-aligned)
#define VP_LD 40   // padded halves per V-panel row (80B, 16B-aligned)

__global__ void __launch_bounds__(256)
k_attn(const _Float16* __restrict__ Qh, const _Float16* __restrict__ Kh,
       const _Float16* __restrict__ Vt, _Float16* __restrict__ Ah) {
  __shared__ _Float16 Ksm[2][32 * KP_LD];
  __shared__ _Float16 Vsm[2][64 * VP_LD];
  __shared__ _Float16 Psm[8][16 * 32];       // per-wave P staging

  const int tid = threadIdx.x;
  const int lane = tid & 31, wslot = tid >> 5;
  const int bh = blockIdx.x >> 4;                       // block-uniform (b,h)
  const int qt = ((blockIdx.x & 15) << 3) + wslot;      // 0..127
  const int bi = bh >> 4, hh = bh & 15;

  const _Float16* Qbase = Qh + ((size_t)bh * SEQ + qt * 16) * HD;
  const _Float16* Kbase = Kh + (size_t)bh * SEQ * HD;
  const _Float16* Vbase = Vt + (size_t)bh * HD * SEQ;
  _Float16* P = &Psm[wslot][0];

  // staging indices: K panel 32x64 halves, V panel 64x32 halves
  const int kr = tid >> 3, kc = (tid & 7) * 8;          // 32 rows x 8 chunks
  const int vr = tid >> 2, vc = (tid & 3) * 8;          // 64 rows x 4 chunks

  const v16h aq0 = load_a_frag(Qbase, HD, 0, lane);
  const v16h aq1 = load_a_frag(Qbase, HD, 32, lane);

  v8f o[4] = {};
  float mrow[8], lrow[8];
#pragma unroll
  for (int j = 0; j < 8; ++j) { mrow[j] = -1.0e30f; lrow[j] = 0.0f; }
  const float scale = 0.125f;                // 1/sqrt(64)
  const int hf = lane >> 4, cn = lane & 15;

  // stage first key block
  async_g2l_b128(Kbase + (size_t)kr * HD + kc, &Ksm[0][kr * KP_LD + kc]);
  async_g2l_b128(Vbase + (size_t)vr * SEQ + vc, &Vsm[0][vr * VP_LD + vc]);
  wait_async0();
  __syncthreads();

  for (int j0 = 0; j0 < SEQ; j0 += 32) {
    const int cur = (j0 >> 5) & 1;
    if (j0 + 32 < SEQ) {                       // prefetch next K/V panels
      async_g2l_b128(Kbase + (size_t)(j0 + 32 + kr) * HD + kc,
                     &Ksm[cur ^ 1][kr * KP_LD + kc]);
      async_g2l_b128(Vbase + (size_t)vr * SEQ + (j0 + 32) + vc,
                     &Vsm[cur ^ 1][vr * VP_LD + vc]);
    }
    const _Float16* Kb = &Ksm[cur][0];
    const _Float16* Vb = &Vsm[cur][0];

    // scores for keys [j0, j0+16) and [j0+16, j0+32)
    v8f s0 = {}, s1 = {};
    s0 = wmma_f16(aq0, lds_b_frag16(Kb, 0, KP_LD, lane), s0);
    s0 = wmma_f16(aq1, lds_b_frag16(Kb + 32, 0, KP_LD, lane), s0);
    s1 = wmma_f16(aq0, lds_b_frag16(Kb, 1, KP_LD, lane), s1);
    s1 = wmma_f16(aq1, lds_b_frag16(Kb + 32, 1, KP_LD, lane), s1);

#pragma unroll
    for (int j = 0; j < 8; ++j) {
      float a0 = s0[j] * scale, a1 = s1[j] * scale;
      float rm = fmaxf(a0, a1);
#pragma unroll
      for (int off = 1; off < 16; off <<= 1)
        rm = fmaxf(rm, __shfl_xor(rm, off, 32));
      float mnew  = fmaxf(mrow[j], rm);
      float alpha = __expf(mrow[j] - mnew);
      float p0 = __expf(a0 - mnew), p1 = __expf(a1 - mnew);
      mrow[j] = mnew;
      float rs = p0 + p1;
#pragma unroll
      for (int off = 1; off < 16; off <<= 1)
        rs += __shfl_xor(rs, off, 32);
      lrow[j] = lrow[j] * alpha + rs;
#pragma unroll
      for (int t = 0; t < 4; ++t) o[t][j] *= alpha;
      // stage P (row-major 16x32) for re-layout into A fragment
      int r = j + 8 * hf;
      P[r * 32 + cn]      = (_Float16)p0;
      P[r * 32 + 16 + cn] = (_Float16)p1;
    }
    asm volatile("s_wait_dscnt 0x0" ::: "memory");

    // reload P in A-fragment layout
    v16h ap;
    {
      const _Float16* pp = P + (lane & 15) * 32 + 16 * hf;
      v8h lo = *(const v8h*)pp;
      v8h hi = *(const v8h*)(pp + 8);
#pragma unroll
      for (int i = 0; i < 8; ++i) { ap[i] = lo[i]; ap[i + 8] = hi[i]; }
    }
    // O += P @ V  (V panel rows = head-dim, cols = keys of this block)
#pragma unroll
    for (int t = 0; t < 4; ++t)
      o[t] = wmma_f16(ap, lds_b_frag16(Vb, t, VP_LD, lane), o[t]);

    wait_async0();
    __syncthreads();
  }

  // normalize and write attention output as f16 [B, S, D_MODEL]
#pragma unroll
  for (int j = 0; j < 8; ++j) {
    int s = qt * 16 + j + 8 * hf;
    float inv = 1.0f / lrow[j];
#pragma unroll
    for (int t = 0; t < 4; ++t) {
      Ah[((size_t)bi * SEQ + s) * D_MODEL + hh * 64 + t * 16 + cn] =
          (_Float16)(o[t][j] * inv);
    }
  }
}

// ---------------------------------------------------------------------------
// GEMM2: out = attn_out @ W_out + b_out (f32 output). Same blocking as GEMM1.
// ---------------------------------------------------------------------------
__global__ void __launch_bounds__(256)
k_gemm_out(const _Float16* __restrict__ Ah, const _Float16* __restrict__ WoT,
           const float* __restrict__ bout, float* __restrict__ out) {
  __shared__ _Float16 Bsm[2][64 * BP_LD];
  const int tid = threadIdx.x;
  const int lane = tid & 31, wslot = tid >> 5;
  const int NT = D_MODEL / 64;               // 16 col tiles
  const int bm = blockIdx.x / NT;
  const int bn = blockIdx.x - bm * NT;
  const int row0 = bm * 128 + wslot * 16;
  const int col0 = bn * 64;

  const _Float16* gB = WoT + (size_t)col0 * D_MODEL;
  const int sr = tid >> 2, sc = (tid & 3) * 8;

  v8f acc[4] = {};
  const _Float16* Arow = Ah + (size_t)row0 * D_MODEL;

  async_g2l_b128(gB + (size_t)sr * D_MODEL + sc, &Bsm[0][sr * BP_LD + sc]);
  wait_async0();
  __syncthreads();

  for (int k0 = 0; k0 < D_MODEL; k0 += 32) {
    const int cur = (k0 >> 5) & 1;
    if (k0 + 32 < D_MODEL)
      async_g2l_b128(gB + (size_t)sr * D_MODEL + (k0 + 32) + sc,
                     &Bsm[cur ^ 1][sr * BP_LD + sc]);
    v16h a = load_a_frag(Arow, D_MODEL, k0, lane);
    const _Float16* Bb = &Bsm[cur][0];
#pragma unroll
    for (int t = 0; t < 4; ++t)
      acc[t] = wmma_f16(a, lds_b_frag32(Bb, t, BP_LD, lane), acc[t]);
    wait_async0();
    __syncthreads();
  }

  const int hf = lane >> 4, n = lane & 15;
#pragma unroll
  for (int t = 0; t < 4; ++t) {
#pragma unroll
    for (int j = 0; j < 8; ++j) {
      int row = row0 + j + 8 * hf;
      int col = col0 + t * 16 + n;
      out[(size_t)row * D_MODEL + col] = acc[t][j] + bout[col];
    }
  }
}

// ---------------------------------------------------------------------------
// Host-side launcher
// ---------------------------------------------------------------------------
extern "C" void kernel_launch(void* const* d_in, const int* in_sizes, int n_in,
                              void* d_out, int out_size, void* d_ws,
                              size_t ws_size, hipStream_t stream) {
  const float* x    = (const float*)d_in[0];
  const float* Wqkv = (const float*)d_in[1];
  const float* bqkv = (const float*)d_in[2];
  const float* Wout = (const float*)d_in[3];
  const float* bout = (const float*)d_in[4];
  float* out = (float*)d_out;

  char* ws = (char*)d_ws;
  const size_t MB = 1024 * 1024;
  _Float16* xh    = (_Float16*)(ws + 0);          // 16 MB: x in f16
  _Float16* WqkvT = (_Float16*)(ws + 16 * MB);    //  6 MB: W_qkv^T f16
  _Float16* WoutT = (_Float16*)(ws + 24 * MB);    //  2 MB: W_out^T f16
  _Float16* Qh    = (_Float16*)(ws + 26 * MB);    // 16 MB: Q [B,H,S,64]
  _Float16* Kh    = (_Float16*)(ws + 42 * MB);    // 16 MB: K [B,H,S,64]
  _Float16* Vt    = (_Float16*)(ws + 58 * MB);    // 16 MB: V [B,H,64,S]
  _Float16* Ah    = (_Float16*)(ws + 74 * MB);    // 16 MB: attn out [B,S,D]

  // 1) precision conversion + weight transposes
  k_cvt_f32_f16<<<(MTOT * D_MODEL / 4 + 255) / 256, 256, 0, stream>>>(
      x, xh, MTOT * D_MODEL);
  k_cvt_transpose<<<(D_MODEL * NQKV + 255) / 256, 256, 0, stream>>>(
      Wqkv, WqkvT, D_MODEL, NQKV);
  k_cvt_transpose<<<(D_MODEL * D_MODEL + 255) / 256, 256, 0, stream>>>(
      Wout, WoutT, D_MODEL, D_MODEL);

  // 2) QKV projection: 64 row-blocks x 48 col-tiles
  k_gemm_qkv<<<64 * 48, 256, 0, stream>>>(xh, WqkvT, bqkv, Qh, Kh, Vt);

  // 3) attention: 64 (b,h) groups x 16 blocks of 8 query-tile waves
  k_attn<<<64 * 16, 256, 0, stream>>>(Qh, Kh, Vt, Ah);

  // 4) output projection: 64 row-blocks x 16 col-tiles
  k_gemm_out<<<64 * 16, 256, 0, stream>>>(Ah, WoutT, bout, out);
}
